// CrossMultiHeadedSelfAttention_22179211116652
// MI455X (gfx1250) — matrix operand order repaired and checked by reading, not compiled, not run on hardware
//
#include <hip/hip_runtime.h>

// ---------------------------------------------------------------- types
typedef __attribute__((ext_vector_type(16))) __bf16          v16bf;
typedef __attribute__((ext_vector_type(16))) unsigned short  v16u;
typedef __attribute__((ext_vector_type(8)))  unsigned short  v8u;
typedef __attribute__((ext_vector_type(8)))  float           v8f;

static __device__ __forceinline__ unsigned short f2bf(float f) {
  unsigned u = __float_as_uint(f);
  u += 0x7fffu + ((u >> 16) & 1u);            // round-to-nearest-even
  return (unsigned short)(u >> 16);
}
static __device__ __forceinline__ v16u cat8(v8u lo, v8u hi) {
  return __builtin_shufflevector(lo, hi, 0,1,2,3,4,5,6,7,8,9,10,11,12,13,14,15);
}
static __device__ __forceinline__ v8f wmma_bf16(v16u a, v16u b, v8f c) {
  return __builtin_amdgcn_wmma_f32_16x16x32_bf16(
      false, __builtin_bit_cast(v16bf, a),
      false, __builtin_bit_cast(v16bf, b),
      (short)0, c, false, false);
}
// lane <-> lane^16 exchange: ds_swizzle group-of-32, xor=0x10, and=0x1f
static __device__ __forceinline__ float swapx16(float v) {
  return __int_as_float(__builtin_amdgcn_ds_swizzle(__float_as_int(v), 0x401f));
}
// gfx1250 async global->LDS copy (ASYNCcnt); lds generic ptr low 32 bits are
// the LDS offset (shared aperture lives in addr[63:32]).
static __device__ __forceinline__ void async_ld_b128(void* lds, const void* gptr) {
  unsigned off = (unsigned)(size_t)lds;
  asm volatile("global_load_async_to_lds_b128 %0, %1, off"
               :: "v"(off), "v"(gptr) : "memory");
}
static __device__ __forceinline__ void wait_async0() {
  asm volatile("s_wait_asynccnt 0x0" ::: "memory");
}

// problem constants
#define BB  2
#define SQ  2048
#define SK  2048
#define DD  1024
#define HH  16
#define HD  64
#define MROWS (BB * SQ)   // 4096

// ---------------------------------------------------------------- prep: f32 -> bf16
__global__ __launch_bounds__(256) void k_cvt_bf16(const float* __restrict__ in,
                                                  unsigned short* __restrict__ out,
                                                  int n4) {
  int i = blockIdx.x * 256 + threadIdx.x;
  if (i >= n4) return;
  float4 f = ((const float4*)in)[i];
  uint2 o;
  o.x = (unsigned)f2bf(f.x) | ((unsigned)f2bf(f.y) << 16);
  o.y = (unsigned)f2bf(f.z) | ((unsigned)f2bf(f.w) << 16);
  ((uint2*)out)[i] = o;
}

// ---------------------------------------------------------------- prep: transpose+convert weights
// MODE 0: W is [H][D][HD] f32, out T[n][k] bf16 with n=h*64+e, k=d
// MODE 1: W is [K][N] f32 (Wo), out T[n][k] bf16
template <int MODE>
__global__ __launch_bounds__(256) void k_cvt_w(const float* __restrict__ W0,
                                               const float* __restrict__ W1,
                                               const float* __restrict__ W2,
                                               unsigned short* __restrict__ T0,
                                               unsigned short* __restrict__ T1,
                                               unsigned short* __restrict__ T2) {
  __shared__ float Ts[64][65];
  const float* W = (blockIdx.z == 0) ? W0 : (blockIdx.z == 1 ? W1 : W2);
  unsigned short* T = (blockIdx.z == 0) ? T0 : (blockIdx.z == 1 ? T1 : T2);
  const int k0 = blockIdx.x * 64, n0 = blockIdx.y * 64;
  const int tid = threadIdx.x;
#pragma unroll
  for (int i = 0; i < 4; ++i) {
    int c = tid + i * 256;              // 1024 float4 chunks over [64 k][64 e]
    int kk = c >> 4, e = (c & 15) * 4;
    const float* src = (MODE == 0)
        ? W + ((size_t)(n0 >> 6) * DD + (k0 + kk)) * HD + e
        : W + (size_t)(k0 + kk) * DD + n0 + e;
    float4 f = *(const float4*)src;
    Ts[kk][e + 0] = f.x; Ts[kk][e + 1] = f.y;
    Ts[kk][e + 2] = f.z; Ts[kk][e + 3] = f.w;
  }
  __syncthreads();
#pragma unroll
  for (int i = 0; i < 8; ++i) {
    int c = tid + i * 256;              // 2048 uint chunks over [64 e][32 kpair]
    int e = c >> 5, kp = (c & 31) * 2;
    unsigned pk = (unsigned)f2bf(Ts[kp][e]) | ((unsigned)f2bf(Ts[kp + 1][e]) << 16);
    *(unsigned int*)&T[(size_t)(n0 + e) * DD + k0 + kp] = pk;
  }
}

// ---------------------------------------------------------------- bf16 GEMM, C = A * Bt^T + bias
// A: [4096][1024] bf16 row-major;  Bt: [1024 n][1024 k] bf16
// Double-buffered with async global->LDS staging (ASYNCcnt).
// OUTMODE 0: f32 row-major [4096][1024];  OUTMODE 1: bf16 [B][H][S][HD]
template <int OUTMODE>
__global__ __launch_bounds__(256) void k_gemm(const unsigned short* __restrict__ A,
                                              const unsigned short* __restrict__ Bt,
                                              const float* __restrict__ bias,
                                              void* __restrict__ Cout) {
  __shared__ unsigned short As[2][128][32];
  __shared__ unsigned short Bs[2][128][32];
  const int tid = threadIdx.x, w = tid >> 5, l = tid & 31;
  const int lq = l & 15, lh = l >> 4;
  const int m0 = blockIdx.y * 128, n0 = blockIdx.x * 128;
  const int wm = (w >> 2) * 64, wn = (w & 3) * 32;

  // per-thread staging coordinates (2 x 16B chunks each of A and B per tile)
  const int c0r = tid >> 2,        c0k = (tid & 3) * 8;
  const int c1r = (tid + 256) >> 2, c1k = ((tid + 256) & 3) * 8;
  const unsigned short* Arow0 = A  + (size_t)(m0 + c0r) * DD + c0k;
  const unsigned short* Arow1 = A  + (size_t)(m0 + c1r) * DD + c1k;
  const unsigned short* Brow0 = Bt + (size_t)(n0 + c0r) * DD + c0k;
  const unsigned short* Brow1 = Bt + (size_t)(n0 + c1r) * DD + c1k;

  v8f acc[4][2];
#pragma unroll
  for (int mt = 0; mt < 4; ++mt)
#pragma unroll
    for (int nt = 0; nt < 2; ++nt)
#pragma unroll
      for (int r = 0; r < 8; ++r) acc[mt][nt][r] = 0.f;

  auto stage = [&](int ko, int buf) {
    const int k0 = ko * 32;
    async_ld_b128(&As[buf][c0r][c0k], Arow0 + k0);
    async_ld_b128(&As[buf][c1r][c1k], Arow1 + k0);
    async_ld_b128(&Bs[buf][c0r][c0k], Brow0 + k0);
    async_ld_b128(&Bs[buf][c1r][c1k], Brow1 + k0);
  };
  auto compute = [&](const unsigned short (&AsB)[128][32],
                     const unsigned short (&BsB)[128][32]) {
    v16u bfr[2];
#pragma unroll
    for (int nt = 0; nt < 2; ++nt) {
      const unsigned short* br = &BsB[wn + nt * 16 + lq][lh * 8];
      bfr[nt] = cat8(*(const v8u*)br, *(const v8u*)(br + 16));
    }
#pragma unroll
    for (int mt = 0; mt < 4; ++mt) {
      const unsigned short* ar = &AsB[wm + mt * 16 + lq][lh * 8];
      v16u af = cat8(*(const v8u*)ar, *(const v8u*)(ar + 16));
#pragma unroll
      for (int nt = 0; nt < 2; ++nt)
        acc[mt][nt] = wmma_bf16(af, bfr[nt], acc[mt][nt]);
    }
  };

  stage(0, 0);
  wait_async0();
  __syncthreads();
  for (int ko = 0; ko < DD / 32; ko += 2) {
    if (ko + 1 < DD / 32) stage(ko + 1, 1);   // prefetch into buf1
    compute(As[0], Bs[0]);
    wait_async0();
    __syncthreads();
    if (ko + 2 < DD / 32) stage(ko + 2, 0);   // prefetch into buf0
    compute(As[1], Bs[1]);
    wait_async0();
    __syncthreads();
  }

#pragma unroll
  for (int nt = 0; nt < 2; ++nt) {
    const int n = n0 + wn + nt * 16 + lq;
    const float bv = bias[n];
#pragma unroll
    for (int mt = 0; mt < 4; ++mt) {
      const int mb = m0 + wm + mt * 16 + lh * 8;
      if (OUTMODE == 0) {
        float* C = (float*)Cout;
#pragma unroll
        for (int r = 0; r < 8; ++r)
          C[(size_t)(mb + r) * DD + n] = acc[mt][nt][r] + bv;
      } else {
        unsigned short* C = (unsigned short*)Cout;
        const int hh = n >> 6, e = n & 63;
#pragma unroll
        for (int r = 0; r < 8; ++r) {
          const int m = mb + r;
          const int b = m >> 11, s = m & (SQ - 1);
          C[(((size_t)(b * HH + hh) * SQ) + s) * HD + e] = f2bf(acc[mt][nt][r] + bv);
        }
      }
    }
  }
}

// ---------------------------------------------------------------- flash attention
// Q,K,V: bf16 [B][H][S][64]; O: bf16 [B][S][H*64]
__global__ __launch_bounds__(256) void k_attn(const unsigned short* __restrict__ Qg,
                                              const unsigned short* __restrict__ Kg,
                                              const unsigned short* __restrict__ Vg,
                                              unsigned short* __restrict__ Og) {
  __shared__ unsigned short Qs[128][64];      // scaled Q rows
  __shared__ unsigned short Ks[64][64];       // K rows (async-staged)
  __shared__ unsigned short Vts[64][66];      // V transposed [hd][key], padded
  __shared__ unsigned short Pts[8][16][64];   // per-wave P^T as [q][key]

  const int tid = threadIdx.x, w = tid >> 5, l = tid & 31;
  const int lq = l & 15, lh = l >> 4;
  const int qb = blockIdx.x, h = blockIdx.y, b = blockIdx.z;

  const size_t bh = (size_t)(b * HH + h);
  const unsigned short* Qbase = Qg + (bh * SQ + (size_t)qb * 128) * HD;
  const unsigned short* Kbase = Kg + bh * SK * HD;
  const unsigned short* Vbase = Vg + bh * SK * HD;

  // stage Q, folding in 1/sqrt(HD)=0.125 (exact for bf16)
#pragma unroll
  for (int i = 0; i < 4; ++i) {
    int c = tid + i * 256;                    // 1024 v8u chunks
    int row = c >> 3, e0 = (c & 7) * 8;
    v8u qv = *(const v8u*)(Qbase + (size_t)row * HD + e0);
    v8u sv;
#pragma unroll
    for (int j = 0; j < 8; ++j) {
      float f = __uint_as_float(((unsigned)qv[j]) << 16) * 0.125f;
      sv[j] = (unsigned short)(__float_as_uint(f) >> 16);
    }
    *(v8u*)&Qs[row][e0] = sv;
  }
  __syncthreads();

  // hoist Q fragments (B operand of S^T = K * Q^T), constant over key loop
  v16u qf[2];
#pragma unroll
  for (int kki = 0; kki < 2; ++kki) {
    const unsigned short* qr = &Qs[w * 16 + lq][kki * 32 + lh * 8];
    qf[kki] = cat8(*(const v8u*)qr, *(const v8u*)(qr + 16));
  }

  v8f ot[4];
#pragma unroll
  for (int t = 0; t < 4; ++t)
#pragma unroll
    for (int r = 0; r < 8; ++r) ot[t][r] = 0.f;
  float mrun = -3.0e38f, lrun = 0.f;

  // per-thread staging coordinates
  const int kr0 = tid >> 3,          ke0 = (tid & 7) * 8;          // K chunk 0
  const int kr1 = (tid + 256) >> 3,  ke1 = ((tid + 256) & 7) * 8;  // K chunk 1
  const int vrp = tid >> 3,          ve0 = (tid & 7) * 8;          // V row pair

  for (int kt = 0; kt < SK / 64; ++kt) {
    __syncthreads();                          // previous tile fully consumed
    const unsigned short* Kt = Kbase + (size_t)kt * 64 * HD;
    const unsigned short* Vt = Vbase + (size_t)kt * 64 * HD;
    // K tile: async global->LDS (ASYNCcnt)
    async_ld_b128(&Ks[kr0][ke0], Kt + (size_t)kr0 * HD + ke0);
    async_ld_b128(&Ks[kr1][ke1], Kt + (size_t)kr1 * HD + ke1);
    {                                         // V tile transposed, pair-packed stores
      v8u a = *(const v8u*)(Vt + (size_t)(2 * vrp) * HD + ve0);
      v8u c = *(const v8u*)(Vt + (size_t)(2 * vrp + 1) * HD + ve0);
#pragma unroll
      for (int j = 0; j < 8; ++j)
        *(unsigned int*)&Vts[ve0 + j][2 * vrp] = (unsigned)a[j] | ((unsigned)c[j] << 16);
    }
    wait_async0();
    __syncthreads();

    // S^T (keys x q): A = K rows, B = Q^T columns
    v8f st[4];
#pragma unroll
    for (int m = 0; m < 4; ++m)
#pragma unroll
      for (int r = 0; r < 8; ++r) st[m][r] = 0.f;
#pragma unroll
    for (int kki = 0; kki < 2; ++kki) {
      const int kk = kki * 32;
#pragma unroll
      for (int m = 0; m < 4; ++m) {
        const unsigned short* kr = &Ks[m * 16 + lq][kk + lh * 8];
        v16u af = cat8(*(const v8u*)kr, *(const v8u*)(kr + 16));
        st[m] = wmma_bf16(af, qf[kki], st[m]);
      }
    }

    // online softmax for query column lq (halves merged via lane^16 swizzle)
    float mloc = -3.0e38f;
#pragma unroll
    for (int m = 0; m < 4; ++m)
#pragma unroll
      for (int r = 0; r < 8; ++r) mloc = fmaxf(mloc, st[m][r]);
    mloc = fmaxf(mloc, swapx16(mloc));
    const float mnew = fmaxf(mrun, mloc);
    const float corr = __expf(mrun - mnew);
    float lloc = 0.f;
#pragma unroll
    for (int m = 0; m < 4; ++m)
#pragma unroll
      for (int r = 0; r < 8; ++r) {
        float p = __expf(st[m][r] - mnew);
        st[m][r] = p;
        lloc += p;
      }
    lrun = lrun * corr + lloc + swapx16(lloc);
    mrun = mnew;
#pragma unroll
    for (int t = 0; t < 4; ++t)
#pragma unroll
      for (int r = 0; r < 8; ++r) ot[t][r] *= corr;

    // P^T -> wave-private LDS [q][key] (same-wave DS ops are in-order)
#pragma unroll
    for (int m = 0; m < 4; ++m)
#pragma unroll
      for (int r = 0; r < 8; r += 2) {
        int key = m * 16 + lh * 8 + r;
        unsigned pk = (unsigned)f2bf(st[m][r]) | ((unsigned)f2bf(st[m][r + 1]) << 16);
        *(unsigned int*)&Pts[w][lq][key] = pk;
      }

    // O^T += V^T * P^T
#pragma unroll
    for (int kki = 0; kki < 2; ++kki) {
      const int kk = kki * 32;
      const unsigned short* pr = &Pts[w][lq][kk + lh * 8];
      v16u pf = cat8(*(const v8u*)pr, *(const v8u*)(pr + 16));
#pragma unroll
      for (int t = 0; t < 4; ++t) {
        const unsigned short* vr = &Vts[t * 16 + lq][kk + lh * 8];
        v8u lo, hi;                           // stride-66 rows: 4B-aligned only
        __builtin_memcpy(&lo, vr, 16);
        __builtin_memcpy(&hi, vr + 16, 16);
        ot[t] = wmma_bf16(cat8(lo, hi), pf, ot[t]);
      }
    }
  }

  const float inv = (lrun > 0.f) ? (1.f / lrun) : 0.f;
  const int qabs = qb * 128 + w * 16 + lq;
  const size_t base = ((size_t)b * SQ + qabs) * DD + h * HD;
#pragma unroll
  for (int t = 0; t < 4; ++t)
#pragma unroll
    for (int r = 0; r < 8; r += 2) {
      int hd = t * 16 + lh * 8 + r;
      unsigned pk = (unsigned)f2bf(ot[t][r] * inv) |
                    ((unsigned)f2bf(ot[t][r + 1] * inv) << 16);
      *(unsigned int*)&Og[base + hd] = pk;
    }
}

// ---------------------------------------------------------------- launch
extern "C" void kernel_launch(void* const* d_in, const int* in_sizes, int n_in,
                              void* d_out, int out_size, void* d_ws, size_t ws_size,
                              hipStream_t stream) {
  const float* x   = (const float*)d_in[0];
  const float* enc = (const float*)d_in[1];
  const float* Wq  = (const float*)d_in[2];
  const float* bq  = (const float*)d_in[3];
  const float* Wk  = (const float*)d_in[4];
  const float* bk  = (const float*)d_in[5];
  const float* Wv  = (const float*)d_in[6];
  const float* bv  = (const float*)d_in[7];
  const float* Wo  = (const float*)d_in[8];
  const float* bo  = (const float*)d_in[9];
  float* out = (float*)d_out;

  char* ws = (char*)d_ws;
  const size_t SZ_ACT = (size_t)MROWS * DD * 2;    // 8 MB
  const size_t SZ_W   = (size_t)DD * DD * 2;       // 2 MB
  unsigned short* xb  = (unsigned short*)(ws);
  unsigned short* eb  = (unsigned short*)(ws + SZ_ACT);
  unsigned short* wqt = (unsigned short*)(ws + 2 * SZ_ACT);
  unsigned short* wkt = (unsigned short*)(ws + 2 * SZ_ACT + SZ_W);
  unsigned short* wvt = (unsigned short*)(ws + 2 * SZ_ACT + 2 * SZ_W);
  unsigned short* wot = (unsigned short*)(ws + 2 * SZ_ACT + 3 * SZ_W);
  unsigned short* qg  = (unsigned short*)(ws + 2 * SZ_ACT + 4 * SZ_W);
  unsigned short* kg  = (unsigned short*)(ws + 3 * SZ_ACT + 4 * SZ_W);
  unsigned short* vg  = (unsigned short*)(ws + 4 * SZ_ACT + 4 * SZ_W);
  unsigned short* oc  = (unsigned short*)(ws + 5 * SZ_ACT + 4 * SZ_W);
  // total workspace used: 6*8MB + 4*2MB = 56 MB

  const int n4 = MROWS * DD / 4;
  k_cvt_bf16<<<n4 / 256, 256, 0, stream>>>(x, xb, n4);
  k_cvt_bf16<<<n4 / 256, 256, 0, stream>>>(enc, eb, n4);
  k_cvt_w<0><<<dim3(DD / 64, DD / 64, 3), 256, 0, stream>>>(Wq, Wk, Wv, wqt, wkt, wvt);
  k_cvt_w<1><<<dim3(DD / 64, DD / 64, 1), 256, 0, stream>>>(Wo, Wo, Wo, wot, wot, wot);

  dim3 gg(DD / 128, MROWS / 128);                  // (8, 32)
  k_gemm<1><<<gg, 256, 0, stream>>>(xb, wqt, bq, qg);
  k_gemm<1><<<gg, 256, 0, stream>>>(eb, wkt, bk, kg);
  k_gemm<1><<<gg, 256, 0, stream>>>(eb, wvt, bv, vg);

  k_attn<<<dim3(SQ / 128, HH, BB), 256, 0, stream>>>(qg, kg, vg, oc);

  k_gemm<0><<<gg, 256, 0, stream>>>(oc, wot, bo, out);
}
